// LinkNet34_26396869001297
// MI455X (gfx1250) — matrix operand order repaired and checked by reading, not compiled, not run on hardware
//
#include <hip/hip_runtime.h>
#include <math.h>

// ---------------------------------------------------------------------------
// CDNA5 (gfx1250) WMMA types
// ---------------------------------------------------------------------------
typedef __attribute__((ext_vector_type(16))) __bf16 v16bf;
typedef __attribute__((ext_vector_type(8)))  float  v8f;

union Frag16 { v16bf v; uint4 u[2]; };

static __device__ __forceinline__ v8f wmma_bf16f32(v16bf a, v16bf b, v8f c) {
  // 8 args: (neg_a, A, neg_b, B, c_mod, C, reuse_a, reuse_b)
  return __builtin_amdgcn_wmma_f32_16x16x32_bf16(false, a, false, b, (short)0, c,
                                                 false, false);
}

// async memory -> LDS copy, 16B per lane, tracked by ASYNCcnt
static __device__ __forceinline__ void async_b128_to_lds(unsigned lds_off,
                                                         const void* gaddr) {
  asm volatile("global_load_async_to_lds_b128 %0, %1, off"
               :: "v"(lds_off), "v"(gaddr) : "memory");
}
static __device__ __forceinline__ void wait_async0() {
  asm volatile("s_wait_asynccnt 0x0" ::: "memory");
}

// ---------------------------------------------------------------------------
// Problem constants (from reference): B=32, C=256, Hk=Wk=8, ws=7
// ---------------------------------------------------------------------------
#define CDIM   256
#define SEQ    50          // 1 cls + 49 patches
#define MCLS   2048        // 32*64 cls tokens
#define MBIG   102400      // 2048 windows * 50 tokens
#define QKVN   768
#define HID    1024

// ---------------------------------------------------------------------------
// f32 -> bf16 conversion
// ---------------------------------------------------------------------------
__global__ void k_f2bf(const float* __restrict__ s, __bf16* __restrict__ d, int n) {
  int i = blockIdx.x * 256 + threadIdx.x;
  if (i < n) d[i] = (__bf16)s[i];
}

// ---------------------------------------------------------------------------
// relative-position bias, padded to (8, 64, 64); zeros outside [1..49]^2
// ---------------------------------------------------------------------------
__global__ void k_bias(const float* __restrict__ rel, float* __restrict__ bias) {
  int i = blockIdx.x * 256 + threadIdx.x;
  if (i >= 8 * 64 * 64) return;
  int n = i & 63, m = (i >> 6) & 63, h = i >> 12;
  float v = 0.f;
  if (m >= 1 && m <= 49 && n >= 1 && n <= 49) {
    int p = m - 1, q = n - 1;
    int yi = p / 7, xi = p % 7, yj = q / 7, xj = q % 7;
    v = rel[h * 169 + (yi - yj + 6) * 13 + (xi - xj + 6)];
  }
  bias[i] = v;
}

// ---------------------------------------------------------------------------
// cls_tokens (B,C,8,8) -> cls_flat (B*64, 256)
// ---------------------------------------------------------------------------
__global__ void k_gather_cls(const float* __restrict__ ct, float* __restrict__ cf) {
  int i = blockIdx.x * 256 + threadIdx.x;  // 32*64*256 exact
  int c = i & 255;
  int n = (i >> 8) & 63;
  int b = i >> 14;
  cf[i] = ct[(size_t)(b * 256 + c) * 64 + n];
}

// ---------------------------------------------------------------------------
// LayerNorm over C=256, one row per block (8 wave32), bf16 output
// ---------------------------------------------------------------------------
__global__ __launch_bounds__(256) void k_ln(const float* __restrict__ x,
                                            const float* __restrict__ wgt,
                                            const float* __restrict__ bst,
                                            __bf16* __restrict__ y) {
  const size_t row = blockIdx.x;
  const int t = threadIdx.x;
  const float v = x[row * 256 + t];
  float s = v, s2 = v * v;
#pragma unroll
  for (int o = 16; o > 0; o >>= 1) {
    s  += __shfl_xor(s,  o, 32);
    s2 += __shfl_xor(s2, o, 32);
  }
  __shared__ float ps[8], ps2[8];
  if ((t & 31) == 0) { ps[t >> 5] = s; ps2[t >> 5] = s2; }
  __syncthreads();
  float tot = 0.f, tot2 = 0.f;
#pragma unroll
  for (int i = 0; i < 8; ++i) { tot += ps[i]; tot2 += ps2[i]; }
  const float mu  = tot * (1.f / 256.f);
  const float var = tot2 * (1.f / 256.f) - mu * mu;
  const float rs  = rsqrtf(var + 1e-5f);
  y[row * 256 + t] = (__bf16)((v - mu) * rs * wgt[t] + bst[t]);
}

// ---------------------------------------------------------------------------
// bf16 WMMA GEMM: C[M,N] = A[M,K] * W[N,K]^T (+bias)(+residual)(+gelu)
// block = 128 threads (4 waves), block tile 128x64, per-wave 32x64, K step 32.
// W tile double-buffered in LDS via global_load_async_to_lds_b128 (ASYNCcnt);
// A streamed from global per lane in native fragment layout.
// outF != null  -> f32 output (optional residual add)
// outF == null  -> bf16 output (optional exact GELU)
// ---------------------------------------------------------------------------
__global__ __launch_bounds__(128) void k_gemm(const __bf16* __restrict__ A,
                                              const __bf16* __restrict__ Bw,
                                              const float* __restrict__ bias,
                                              const float* __restrict__ residual,
                                              float* __restrict__ outF,
                                              __bf16* __restrict__ outB,
                                              int M, int N, int K, int dogelu) {
  __shared__ __bf16 Bs[2][64 * 32];
  const int t = threadIdx.x;
  const int wave = t >> 5;
  const int lane = t & 31;
  const int l = lane & 15;
  const int g = lane >> 4;
  const int tileM = blockIdx.y * 128;
  const int tileN = blockIdx.x * 64;
  const size_t arow0 = (size_t)(tileM + wave * 32 + l) * K;
  const size_t arow1 = arow0 + (size_t)16 * K;

  // staging assignment: thread -> (row, 16-elem half) of the 64x32 W tile
  const int sn  = t >> 1;
  const int shf = (t & 1) * 16;
  const unsigned lds0 = (unsigned)(size_t)(&Bs[0][sn * 32 + shf]);
  const unsigned lds1 = (unsigned)(size_t)(&Bs[1][sn * 32 + shf]);
  const __bf16* srow = Bw + (size_t)(tileN + sn) * K + shf;

  v8f acc[8] = {};

  // prologue: stage k-tile 0 into buffer 0
  async_b128_to_lds(lds0, srow);
  async_b128_to_lds(lds0 + 16, srow + 8);
  wait_async0();
  __syncthreads();

  for (int k0 = 0; k0 < K; k0 += 32) {
    const int cur = (k0 >> 5) & 1;
    if (k0 + 32 < K) {  // stage next k-tile into the other buffer (async)
      const unsigned d = cur ? lds0 : lds1;
      async_b128_to_lds(d, srow + k0 + 32);
      async_b128_to_lds(d + 16, srow + k0 + 40);
    }
    Frag16 a0, a1;  // A frags: lane row, K chunks {k0+8g, k0+16+8g}
    a0.u[0] = *(const uint4*)(A + arow0 + k0 + 8 * g);
    a0.u[1] = *(const uint4*)(A + arow0 + k0 + 16 + 8 * g);
    a1.u[0] = *(const uint4*)(A + arow1 + k0 + 8 * g);
    a1.u[1] = *(const uint4*)(A + arow1 + k0 + 16 + 8 * g);
#pragma unroll
    for (int j = 0; j < 4; ++j) {
      Frag16 b;  // B frag: lane column, K chunks {8g, 16+8g}
      const uint4* bp = (const uint4*)(&Bs[cur][(j * 16 + l) * 32 + 8 * g]);
      b.u[0] = bp[0];
      b.u[1] = bp[2];
      acc[j]     = wmma_bf16f32(a0.v, b.v, acc[j]);
      acc[4 + j] = wmma_bf16f32(a1.v, b.v, acc[4 + j]);
    }
    wait_async0();
    __syncthreads();
  }

#pragma unroll
  for (int half = 0; half < 2; ++half) {
#pragma unroll
    for (int j = 0; j < 4; ++j) {
      const int n = tileN + j * 16 + l;
      const float bv = bias ? bias[n] : 0.f;
#pragma unroll
      for (int r = 0; r < 8; ++r) {
        const int m = tileM + wave * 32 + half * 16 + r + 8 * g;
        float v = acc[half * 4 + j][r] + bv;
        const size_t idx = (size_t)m * N + n;
        if (outF) {
          if (residual) v += residual[idx];
          outF[idx] = v;
        } else {
          if (dogelu) v = 0.5f * v * (1.f + erff(v * 0.70710678118654752f));
          outB[idx] = (__bf16)v;
        }
      }
    }
  }
}

// ---------------------------------------------------------------------------
// Fused attention for one (window, head): seq<=64 padded to 64, hd=32.
// block = 128 threads (4 waves). qkv layout: row stride 768 = [q|k|v]x8x32.
//   scores(64x64) = q k^T * scale (+bias)   : 16 WMMA
//   softmax via LDS (rows masked at seq)
//   out(64x32)   = attn v                   : 16 WMMA (vT staged in LDS)
// ---------------------------------------------------------------------------
__global__ __launch_bounds__(128) void k_attn(const __bf16* __restrict__ qkv,
                                              const float* __restrict__ bias,
                                              __bf16* __restrict__ outp, int seq) {
  __shared__ float  sS[64 * 64];
  __shared__ __bf16 sA[64 * 64];
  __shared__ __bf16 sV[32 * 64];
  const int w = blockIdx.x >> 3;
  const int h = blockIdx.x & 7;
  const int t = threadIdx.x;
  const int wave = t >> 5;
  const int lane = t & 31;
  const int l = lane & 15;
  const int g = lane >> 4;
  const size_t rowbase = (size_t)w * seq;
  const int qoff = h * 32, koff = 256 + h * 32, voff = 512 + h * 32;

  // stage V transposed: sV[d][n], zero-padded for n >= seq
  for (int e = t; e < 32 * 64; e += 128) {
    const int d = e >> 6, n = e & 63;
    sV[d * 64 + n] = (n < seq) ? qkv[(rowbase + n) * 768 + voff + d] : (__bf16)0.f;
  }

  // ---- scores = q @ k^T ----
  const int m0 = wave * 16 + l;
  const int mc = (m0 < seq) ? m0 : seq - 1;
  Frag16 a;
  a.u[0] = *(const uint4*)(qkv + (rowbase + mc) * 768 + qoff + 8 * g);
  a.u[1] = *(const uint4*)(qkv + (rowbase + mc) * 768 + qoff + 16 + 8 * g);
  v8f sc[4] = {};
#pragma unroll
  for (int j = 0; j < 4; ++j) {
    const int n  = j * 16 + l;
    const int nc = (n < seq) ? n : seq - 1;
    Frag16 b;
    const uint4* bp = (const uint4*)(qkv + (rowbase + nc) * 768 + koff + 8 * g);
    b.u[0] = bp[0];
    b.u[1] = bp[2];
    sc[j] = wmma_bf16f32(a.v, b.v, sc[j]);
  }
  const float scale = 0.17677669529663687f;  // 1/sqrt(32)
#pragma unroll
  for (int j = 0; j < 4; ++j) {
    const int n = j * 16 + l;
#pragma unroll
    for (int r = 0; r < 8; ++r) {
      const int m = wave * 16 + r + 8 * g;
      float s = sc[j][r] * scale;
      if (bias) s += bias[((size_t)h * 64 + m) * 64 + n];
      sS[m * 64 + n] = s;
    }
  }
  __syncthreads();

  // ---- softmax over n < seq, write bf16 attn (zeros in padding) ----
  if (t < 64) {
    const int m = t;
    float mx = -3.0e38f;
    for (int n = 0; n < seq; ++n) mx = fmaxf(mx, sS[m * 64 + n]);
    float sum = 0.f;
    for (int n = 0; n < seq; ++n) sum += __expf(sS[m * 64 + n] - mx);
    const float inv = 1.f / sum;
    for (int n = 0; n < 64; ++n)
      sA[m * 64 + n] = (n < seq) ? (__bf16)(__expf(sS[m * 64 + n] - mx) * inv)
                                 : (__bf16)0.f;
  }
  __syncthreads();

  // ---- out = attn @ v ----
  v8f o[2] = {};
#pragma unroll
  for (int kk = 0; kk < 64; kk += 32) {
    Frag16 aa;
    aa.u[0] = *(const uint4*)(&sA[(wave * 16 + l) * 64 + kk + 8 * g]);
    aa.u[1] = *(const uint4*)(&sA[(wave * 16 + l) * 64 + kk + 16 + 8 * g]);
#pragma unroll
    for (int jt = 0; jt < 2; ++jt) {
      Frag16 b;
      const uint4* bp = (const uint4*)(&sV[(jt * 16 + l) * 64 + kk + 8 * g]);
      b.u[0] = bp[0];
      b.u[1] = bp[2];
      o[jt] = wmma_bf16f32(aa.v, b.v, o[jt]);
    }
  }
#pragma unroll
  for (int jt = 0; jt < 2; ++jt) {
#pragma unroll
    for (int r = 0; r < 8; ++r) {
      const int m = wave * 16 + r + 8 * g;
      if (m < seq)
        outp[(rowbase + m) * 256 + h * 32 + jt * 16 + l] = (__bf16)o[jt][r];
    }
  }
}

// ---------------------------------------------------------------------------
// Assemble x (2048, 50, 256): token0 = cls2, tokens 1..49 from patch windows
// ---------------------------------------------------------------------------
__global__ void k_assemble(const float* __restrict__ cls2,
                           const float* __restrict__ patch,
                           float* __restrict__ x) {
  size_t i = (size_t)blockIdx.x * 256 + threadIdx.x;  // 102400*256 exact
  int c = (int)(i & 255);
  size_t j = i >> 8;
  int tkn = (int)(j % 50);
  int w   = (int)(j / 50);
  float v;
  if (tkn == 0) {
    v = cls2[(size_t)w * 256 + c];
  } else {
    int b = w >> 6, win = w & 63, hk = win >> 3, wk = win & 7;
    int p = tkn - 1, ii = p / 7, jj = p % 7;
    v = patch[((size_t)(b * 256 + c) * 56 + hk * 7 + ii) * 56 + wk * 7 + jj];
  }
  x[i] = v;
}

// ---------------------------------------------------------------------------
// Scatter tmp (2048,50,256) -> [cls_out (B,C,8,8) | patch_out (B,C,56,56)]
// ---------------------------------------------------------------------------
__global__ void k_scatter(const float* __restrict__ tmp, float* __restrict__ outp) {
  size_t i = (size_t)blockIdx.x * 256 + threadIdx.x;
  int c = (int)(i & 255);
  size_t j = i >> 8;
  int tkn = (int)(j % 50);
  int w   = (int)(j / 50);
  int b = w >> 6, win = w & 63, hk = win >> 3, wk = win & 7;
  float v = tmp[i];
  if (tkn == 0) {
    outp[((size_t)(b * 256 + c) * 8 + hk) * 8 + wk] = v;
  } else {
    int p = tkn - 1, ii = p / 7, jj = p % 7;
    outp[524288 + ((size_t)(b * 256 + c) * 56 + hk * 7 + ii) * 56 + wk * 7 + jj] = v;
  }
}

// ---------------------------------------------------------------------------
// Host orchestration
// ---------------------------------------------------------------------------
extern "C" void kernel_launch(void* const* d_in, const int* in_sizes, int n_in,
                              void* d_out, int out_size, void* d_ws, size_t ws_size,
                              hipStream_t stream) {
  (void)in_sizes; (void)n_in; (void)out_size; (void)ws_size;
  const float* cls_tokens   = (const float*)d_in[0];
  const float* patch_tokens = (const float*)d_in[1];
  const float* norm0_w = (const float*)d_in[2];
  const float* norm0_b = (const float*)d_in[3];
  const float* norm1_w = (const float*)d_in[4];
  const float* norm1_b = (const float*)d_in[5];
  const float* norm2_w = (const float*)d_in[6];
  const float* norm2_b = (const float*)d_in[7];
  const float* qkv_w   = (const float*)d_in[8];
  const float* proj_w  = (const float*)d_in[9];
  const float* proj_b  = (const float*)d_in[10];
  const float* rel_pos = (const float*)d_in[11];
  const float* fc1_w   = (const float*)d_in[12];
  const float* fc1_b   = (const float*)d_in[13];
  const float* fc2_w   = (const float*)d_in[14];
  const float* fc2_b   = (const float*)d_in[15];
  float* out = (float*)d_out;

  char* ws = (char*)d_ws;
  size_t off = 0;
  auto alloc = [&](size_t bytes) -> char* {
    char* p = ws + off;
    off += (bytes + 255) & ~(size_t)255;
    return p;
  };
  __bf16* wq     = (__bf16*)alloc((size_t)QKVN * CDIM * 2);
  __bf16* wproj  = (__bf16*)alloc((size_t)CDIM * CDIM * 2);
  __bf16* wfc1   = (__bf16*)alloc((size_t)HID * CDIM * 2);
  __bf16* wfc2   = (__bf16*)alloc((size_t)CDIM * HID * 2);
  float*  biasb  = (float*)alloc((size_t)8 * 64 * 64 * 4);
  float*  clsf   = (float*)alloc((size_t)MCLS * CDIM * 4);
  __bf16* clsln  = (__bf16*)alloc((size_t)MCLS * CDIM * 2);
  __bf16* qkvc   = (__bf16*)alloc((size_t)MCLS * QKVN * 2);
  __bf16* attc   = (__bf16*)alloc((size_t)MCLS * CDIM * 2);
  float*  cls2   = (float*)alloc((size_t)MCLS * CDIM * 4);
  float*  xbuf   = (float*)alloc((size_t)MBIG * CDIM * 4);
  __bf16* lnbig  = (__bf16*)alloc((size_t)MBIG * CDIM * 2);
  __bf16* qkvm   = (__bf16*)alloc((size_t)MBIG * QKVN * 2);
  __bf16* attm   = (__bf16*)alloc((size_t)MBIG * CDIM * 2);
  float*  tmp    = (float*)alloc((size_t)MBIG * CDIM * 4);
  __bf16* hbuf   = (__bf16*)alloc((size_t)MBIG * HID * 2);

  // weight conversion + bias table
  k_f2bf<<<QKVN * CDIM / 256, 256, 0, stream>>>(qkv_w, wq, QKVN * CDIM);
  k_f2bf<<<CDIM * CDIM / 256, 256, 0, stream>>>(proj_w, wproj, CDIM * CDIM);
  k_f2bf<<<HID * CDIM / 256, 256, 0, stream>>>(fc1_w, wfc1, HID * CDIM);
  k_f2bf<<<CDIM * HID / 256, 256, 0, stream>>>(fc2_w, wfc2, CDIM * HID);
  k_bias<<<(8 * 64 * 64) / 256, 256, 0, stream>>>(rel_pos, biasb);

  // ---- cls branch: cls2 = cls_flat + proj(attn(LN0(cls_flat))) ----
  k_gather_cls<<<MCLS * CDIM / 256, 256, 0, stream>>>(cls_tokens, clsf);
  k_ln<<<MCLS, 256, 0, stream>>>(clsf, norm0_w, norm0_b, clsln);
  k_gemm<<<dim3(QKVN / 64, MCLS / 128), 128, 0, stream>>>(
      clsln, wq, nullptr, nullptr, nullptr, qkvc, MCLS, QKVN, CDIM, 0);
  k_attn<<<32 * 8, 128, 0, stream>>>(qkvc, nullptr, attc, 64);
  k_gemm<<<dim3(CDIM / 64, MCLS / 128), 128, 0, stream>>>(
      attc, wproj, proj_b, clsf, cls2, nullptr, MCLS, CDIM, CDIM, 0);

  // ---- main branch ----
  k_assemble<<<MBIG * CDIM / 256, 256, 0, stream>>>(cls2, patch_tokens, xbuf);
  k_ln<<<MBIG, 256, 0, stream>>>(xbuf, norm1_w, norm1_b, lnbig);
  k_gemm<<<dim3(QKVN / 64, MBIG / 128), 128, 0, stream>>>(
      lnbig, wq, nullptr, nullptr, nullptr, qkvm, MBIG, QKVN, CDIM, 0);
  k_attn<<<2048 * 8, 128, 0, stream>>>(qkvm, biasb, attm, SEQ);
  k_gemm<<<dim3(CDIM / 64, MBIG / 128), 128, 0, stream>>>(
      attm, wproj, proj_b, xbuf, tmp, nullptr, MBIG, CDIM, CDIM, 0);

  // ---- MLP: tmp += fc2(gelu(fc1(LN2(tmp)))) ----
  k_ln<<<MBIG, 256, 0, stream>>>(tmp, norm2_w, norm2_b, lnbig);
  k_gemm<<<dim3(HID / 64, MBIG / 128), 128, 0, stream>>>(
      lnbig, wfc1, fc1_b, nullptr, nullptr, hbuf, MBIG, HID, CDIM, 1);
  k_gemm<<<dim3(CDIM / 64, MBIG / 128), 128, 0, stream>>>(
      hbuf, wfc2, fc2_b, tmp, tmp, nullptr, MBIG, CDIM, HID, 0);

  // ---- outputs ----
  k_scatter<<<MBIG * CDIM / 256, 256, 0, stream>>>(tmp, out);
}
